// SoftSplatBaseline_49898930045248
// MI455X (gfx1250) — matrix operand order repaired and checked by reading, not compiled, not run on hardware
//
#include <hip/hip_runtime.h>
#include <hip/hip_bf16.h>
#include <math.h>

// ============================================================================
// MI455X (gfx1250) SoftSplat interpolation pipeline.
//  - All 3x3 convs -> implicit GEMM on v_wmma_f32_16x16x32_f16 (f16 in, f32 acc)
//    with fused pre-PReLU, bias, residual add, post-PReLU.
//  - Block tile = 32 Cout x (4 rows x 16 px): vertical halo sharing + 4x
//    amortization of the weight-fragment staging.
//  - K ordered tap-major; stride is a template param so every LDS fragment
//    address is an immediate ds_load_b128 offset and staging uses only
//    constant-divisor arithmetic.
//  - Softmax splatting -> global_atomic_add_f32 scatter + normalize.
// ============================================================================

typedef __attribute__((ext_vector_type(16))) _Float16 v16h;
typedef __attribute__((ext_vector_type(8)))  float    v8f;

#define CONV_TPB  256
#define COUT_TILE 32
#define CIN_CHUNK 32

// ---------------------------------------------------------------------------
// Implicit-GEMM 3x3 conv. Block = 8 waves = 2(Cout sub-tile) x 4(row).
// Each wave owns a 16(Cout)x16(pix) WMMA tile on one output row.
// K = Cin*9 in 32-channel chunks; K-block kb == 3x3 tap index.
// ---------------------------------------------------------------------------
template <int S>
__global__ __launch_bounds__(CONV_TPB) void k_conv3x3_wmma(
    const float* __restrict__ in, const float* __restrict__ wgt,
    const float* __restrict__ bias, const float* __restrict__ residual,
    const float* __restrict__ alphaPre, const float* __restrict__ alphaPost,
    float* __restrict__ out,
    int N, int Cin, int Hin, int Win, int Cout, int Hout, int Wout)
{
    constexpr int IWt = 15 * S + 3;   // staged patch width  (18 | 33)
    constexpr int RR  = 3 * S + 3;    // staged patch rows   (6  | 9)
    // channel-last input patch: plds[(ry*IWt + xloc)*32 + ci]
    __shared__ alignas(32) _Float16 plds[CIN_CHUNK * RR * IWt];
    // fragment-ready A tiles: wldsA[((msub*9 + kidx)*32 + lane)*16 + e]
    __shared__ alignas(32) _Float16 wldsA[2 * 9 * 32 * 16];

    const int tilesPerRow = (Wout + 15) >> 4;
    const int rowGroups   = (Hout + 3) >> 2;
    int bx = (int)blockIdx.x;
    const int tx = bx % tilesPerRow; bx /= tilesPerRow;
    const int yg = bx % rowGroups;
    const int bn = bx / rowGroups;
    const int co0 = (int)blockIdx.y * COUT_TILE;
    const int x0  = tx << 4;
    const int y0  = yg << 2;

    const int tid  = (int)threadIdx.x;
    const int lane = tid & 31;
    const int wv   = tid >> 5;
    const int msub = wv >> 2;          // 0 | 1 : Cout sub-tile
    const int rr   = wv & 3;           // 0..3 : output row in group

    const int   hasPre = (alphaPre != nullptr);
    const float aPre   = hasPre ? alphaPre[0] : 0.0f;

    // per-lane invariants for the fragment loads
    const int lane16 = lane << 4;
    const int aBase  = msub * 9 * 512;                         // halfs
    const int nn     = lane & 15;
    const int khalf  = (lane >= 16) ? 16 : 0;
    const int bBase  = ((rr * S) * IWt + nn * S) * CIN_CHUNK + khalf;

    v8f acc = {};

    for (int cin0 = 0; cin0 < Cin; cin0 += CIN_CHUNK) {
        const int ccount = (Cin - cin0 < CIN_CHUNK) ? (Cin - cin0) : CIN_CHUNK;
        if (cin0 + CIN_CHUNK < Cin) {  // emits global_prefetch_b8
            __builtin_prefetch(&in[(((size_t)bn * Cin + cin0 + CIN_CHUNK) * Hin
                                    + (size_t)(y0 * S)) * Win], 0, 0);
            __builtin_prefetch(&wgt[(((size_t)co0) * Cin + cin0 + CIN_CHUNK) * 9], 0, 0);
        }

        // ---- stage input patch (fp32 -> f16, fused pre-PReLU), channel-last ----
        constexpr int totalP = CIN_CHUNK * RR * IWt;
        for (int i = tid; i < totalP; i += CONV_TPB) {
            const int xloc = i % IWt;             // constant divisor
            int t = i / IWt;
            const int ry = t % RR;
            const int cc = t / RR;
            float v = 0.0f;
            if (cc < ccount) {
                const int yin = y0 * S + ry - 1;
                const int xin = x0 * S + xloc - 1;
                if (yin >= 0 && yin < Hin && xin >= 0 && xin < Win) {
                    v = in[(((size_t)bn * Cin + cin0 + cc) * Hin + yin) * Win + xin];
                    if (hasPre) v = (v >= 0.0f) ? v : aPre * v;
                }
            }
            plds[(ry * IWt + xloc) * CIN_CHUNK + cc] = (_Float16)v;
        }

        // ---- stage weights directly in WMMA A-fragment layout ----
        // slot i -> (msub_s, kidx, lane_s, e); element ci contiguous to match B:
        // lanes 0-15 hold K {0-15}, lanes 16-31 hold {16-31}.
        const int totalW = 2 * 9 * 32 * 16;   // 9216
        for (int i = tid; i < totalW; i += CONV_TPB) {
            const int e      = i & 15;
            const int lane_s = (i >> 4) & 31;
            int r = i >> 9;
            const int kidx   = r % 9;
            const int msub_s = r / 9;
            const int ev = e >> 1;
            const int ci = ((lane_s >= 16) ? 16 : 0) + e;
            const int co = co0 + (lane_s & 15) + (msub_s << 4);
            float v = 0.0f;
            if (co < Cout && ci < ccount)
                v = wgt[((size_t)co * Cin + cin0 + ci) * 9 + kidx];
            wldsA[i] = (_Float16)v;
        }
        __syncthreads();

        // ---- 9 WMMAs: one per 3x3 tap; operands = contiguous 32B LDS loads
        //      with compile-time immediate offsets ----
        #pragma unroll
        for (int kidx = 0; kidx < 9; ++kidx) {
            constexpr int CK = CIN_CHUNK;
            const int ky = kidx / 3;          // compile-time after unroll
            const int kx = kidx - ky * 3;
            const v16h a = *(const v16h*)&wldsA[aBase + kidx * 512 + lane16];
            const v16h b = *(const v16h*)&plds[(ky * IWt + kx) * CK + bBase];
            acc = __builtin_amdgcn_wmma_f32_16x16x32_f16(
                    false, a, false, b, (short)0, acc, false, false);
        }
        __syncthreads();
    }

    // ---- epilogue: bias + residual + post-PReLU, fp32 store ----
    const int   px      = x0 + nn;
    const int   oy      = y0 + rr;
    const int   hasPost = (alphaPost != nullptr);
    const float aPost   = hasPost ? alphaPost[0] : 0.0f;
    #pragma unroll
    for (int v = 0; v < 8; ++v) {
        const int m  = ((lane >= 16) ? 8 : 0) + v;
        const int co = co0 + (msub << 4) + m;
        if (co < Cout && px < Wout && oy < Hout) {
            const size_t oidx = ((((size_t)bn * Cout) + co) * Hout + oy) * Wout + px;
            float r = acc[v] + bias[co];
            if (residual) r += residual[oidx];
            if (hasPost) r = (r >= 0.0f) ? r : aPost * r;
            out[oidx] = r;
        }
    }
}

// ---------------------------------------------------------------------------
// Elementwise / gather / scatter kernels
// ---------------------------------------------------------------------------
__global__ void k_make_frames(const float* __restrict__ x, float* __restrict__ frames,
                              float* __restrict__ raw, int B, int H, int W)
{
    const size_t total = (size_t)2 * B * 3 * H * W;
    size_t idx = (size_t)blockIdx.x * blockDim.x + threadIdx.x;
    if (idx >= total) return;
    const int xw = (int)(idx % W); size_t t = idx / W;
    const int y  = (int)(t % H);   t /= H;
    const int c  = (int)(t % 3);
    const int n2 = (int)(t / 3);
    const int tt = (n2 >= B) ? 1 : 0;
    const int bb = (n2 >= B) ? n2 - B : n2;
    const float v = x[((((size_t)bb * 3 + c) * 2 + tt) * H + y) * W + xw];
    raw[idx]    = v;
    frames[idx] = v * 2.0f - 1.0f;
}

__global__ void k_backwarp_bd(const float* __restrict__ frames, const float* __restrict__ flow,
                              float* __restrict__ out6, int B, int H, int W)
{
    const int N2 = 2 * B;
    const size_t total = (size_t)N2 * 3 * H * W;
    size_t idx = (size_t)blockIdx.x * blockDim.x + threadIdx.x;
    if (idx >= total) return;
    const int xw = (int)(idx % W); size_t t = idx / W;
    const int y  = (int)(t % H);   t /= H;
    const int c  = (int)(t % 3);
    const int n2 = (int)(t / 3);
    const int srcN = (n2 >= B) ? (n2 - B) : (n2 + B);   // concat([fr1, fr0])
    const float fx = flow[(((size_t)n2 * 2 + 0) * H + y) * W + xw];
    const float fy = flow[(((size_t)n2 * 2 + 1) * H + y) * W + xw];
    const float px = (xw + fx) * (float)(W - 1) / (float)W;
    const float py = (y  + fy) * (float)(H - 1) / (float)H;
    const float x0f = floorf(px), y0f = floorf(py);
    const int x0 = (int)x0f, y0 = (int)y0f;
    const float wx1 = px - x0f, wx0 = 1.0f - wx1;
    const float wy1 = py - y0f, wy0 = 1.0f - wy1;
    const int   xs[4] = {x0, x0 + 1, x0, x0 + 1};
    const int   ys[4] = {y0, y0, y0 + 1, y0 + 1};
    const float wg[4] = {wx0 * wy0, wx1 * wy0, wx0 * wy1, wx1 * wy1};
    float g = 0.0f;
    for (int k = 0; k < 4; ++k)
        if (xs[k] >= 0 && xs[k] < W && ys[k] >= 0 && ys[k] < H)
            g += frames[(((size_t)srcN * 3 + c) * H + ys[k]) * W + xs[k]] * wg[k];
    const float ref = frames[(((size_t)n2 * 3 + c) * H + y) * W + xw];
    out6[(((size_t)n2 * 6 + c)     * H + y) * W + xw] = ref;
    out6[(((size_t)n2 * 6 + 3 + c) * H + y) * W + xw] = -fabsf(g - ref);
}

__global__ void k_zero(float* __restrict__ p, size_t n)
{
    size_t idx = (size_t)blockIdx.x * blockDim.x + threadIdx.x;
    if (idx < n) p[idx] = 0.0f;
}

__global__ void k_softsplat_scatter(const float* __restrict__ feat, const float* __restrict__ flowt,
                                    const float* __restrict__ z, float* __restrict__ acc,
                                    int N, int C, int H, int W)
{
    const size_t total = (size_t)N * H * W;
    size_t idx = (size_t)blockIdx.x * blockDim.x + threadIdx.x;
    if (idx >= total) return;
    const int xw = (int)(idx % W); size_t t = idx / W;
    const int y  = (int)(t % H);
    const int n  = (int)(t / H);
    const size_t HW = (size_t)H * W;
    const float fx = flowt[(((size_t)n * 2 + 0) * H + y) * W + xw];
    const float fy = flowt[(((size_t)n * 2 + 1) * H + y) * W + xw];
    const float ez = expf(z[((size_t)n * H + y) * W + xw]);
    const float px = xw + fx, py = y + fy;
    const float x0f = floorf(px), y0f = floorf(py);
    const int x0 = (int)x0f, y0 = (int)y0f;
    const float wx1 = px - x0f, wx0 = 1.0f - wx1;
    const float wy1 = py - y0f, wy0 = 1.0f - wy1;
    const int   xs[4] = {x0, x0 + 1, x0, x0 + 1};
    const int   ys[4] = {y0, y0, y0 + 1, y0 + 1};
    const float wg[4] = {wx0 * wy0, wx1 * wy0, wx0 * wy1, wx1 * wy1};
    const size_t spix = (size_t)y * W + xw;
    for (int k = 0; k < 4; ++k) {
        const int xi = xs[k], yi = ys[k];
        if (xi >= 0 && xi < W && yi >= 0 && yi < H) {
            const float w_ = wg[k] * ez;
            const size_t dpix = (size_t)yi * W + xi;
            for (int c = 0; c < C; ++c)
                atomicAdd(&acc[((size_t)n * (C + 1) + c) * HW + dpix],
                          feat[((size_t)n * C + c) * HW + spix] * w_);
            atomicAdd(&acc[((size_t)n * (C + 1) + C) * HW + dpix], w_);
        }
    }
}

__global__ void k_softsplat_norm(const float* __restrict__ acc, float* __restrict__ out,
                                 int N, int C, int H, int W)
{
    const size_t HW = (size_t)H * W;
    const size_t total = (size_t)N * C * HW;
    size_t idx = (size_t)blockIdx.x * blockDim.x + threadIdx.x;
    if (idx >= total) return;
    const size_t pix = idx % HW; size_t t = idx / HW;
    const int c = (int)(t % C);
    const int n = (int)(t / C);
    const float d = acc[((size_t)n * (C + 1) + C) * HW + pix] + 1e-7f;
    out[idx] = acc[((size_t)n * (C + 1) + c) * HW + pix] / d;
}

__global__ void k_resize_bilinear(const float* __restrict__ in, float* __restrict__ out,
                                  int N, int C, int Hin, int Win, int Hout, int Wout, float mul)
{
    const size_t total = (size_t)N * C * Hout * Wout;
    size_t idx = (size_t)blockIdx.x * blockDim.x + threadIdx.x;
    if (idx >= total) return;
    const int xw = (int)(idx % Wout); size_t t = idx / Wout;
    const int y  = (int)(t % Hout);   t /= Hout;
    const int c  = (int)(t % C);
    const int n  = (int)(t / C);
    float sx = (xw + 0.5f) * (float)Win / (float)Wout - 0.5f;
    float sy = (y  + 0.5f) * (float)Hin / (float)Hout - 0.5f;
    sx = fminf(fmaxf(sx, 0.0f), (float)(Win - 1));
    sy = fminf(fmaxf(sy, 0.0f), (float)(Hin - 1));
    const int x0 = (int)sx, y0 = (int)sy;
    const int x1 = (x0 + 1 < Win) ? x0 + 1 : Win - 1;
    const int y1 = (y0 + 1 < Hin) ? y0 + 1 : Hin - 1;
    const float ax = sx - (float)x0, ay = sy - (float)y0;
    const float* p = in + (((size_t)n * C + c) * Hin) * Win;
    const float v = (1.0f - ay) * ((1.0f - ax) * p[(size_t)y0 * Win + x0] + ax * p[(size_t)y0 * Win + x1])
                  +         ay  * ((1.0f - ax) * p[(size_t)y1 * Win + x0] + ax * p[(size_t)y1 * Win + x1]);
    out[idx] = v * mul;
}

__global__ void k_interflow(const float* __restrict__ flow_lv,
                            const float* __restrict__ t_p, const float* __restrict__ gamma_p,
                            const int* __restrict__ cH_p, const int* __restrict__ imgH_p,
                            float scale, int b, int H, int W, float* __restrict__ out)
{
    const size_t total = (size_t)2 * b * 2 * H * W;
    size_t idx = (size_t)blockIdx.x * blockDim.x + threadIdx.x;
    if (idx >= total) return;
    const int xw = (int)(idx % W); size_t t = idx / W;
    const int y  = (int)(t % H);   t /= H;
    const int ch = (int)(t % 2);
    const int n2 = (int)(t / 2);
    const float tt = t_p[0], gamma = gamma_p[0];
    const int hfull = (int)((float)imgH_p[0] * scale);
    const int r0    = (int)((float)cH_p[0] * scale);
    const float tau = tt + gamma - gamma / (float)hfull * (float)(r0 + y) + 1e-4f;
    const size_t pix = (size_t)y * W + xw;
    const float fy  = flow_lv[(((size_t)n2 * 2) + 1) * (size_t)H * W + pix];
    const float val = flow_lv[(((size_t)n2 * 2) + ch) * (size_t)H * W + pix];
    float res;
    if (n2 < b) res = tau / (1.0f + gamma * fy / (float)hfull) * val;
    else        res = (1.0f - tau) / (1.0f + gamma * (-fy) / (float)hfull) * val;
    out[idx] = res;
}

__global__ void k_clip01(const float* __restrict__ in, float* __restrict__ out, size_t n)
{
    size_t idx = (size_t)blockIdx.x * blockDim.x + threadIdx.x;
    if (idx < n) {
        const float v = (in[idx] + 1.0f) * 0.5f;
        out[idx] = fminf(fmaxf(v, 0.0f), 1.0f);
    }
}

// ---------------------------------------------------------------------------
// Host orchestration
// ---------------------------------------------------------------------------
extern "C" void kernel_launch(void* const* d_in, const int* in_sizes, int n_in,
                              void* d_out, int out_size, void* d_ws, size_t ws_size,
                              hipStream_t stream)
{
    (void)in_sizes; (void)out_size; (void)ws_size;
    if (n_in < 202) return;   // expects x,flow,t,gamma,cH,img_H + 196 param leaves

    const int B = 2, H = 320, W = 512;
    const int N2 = 2 * B;
    const int H1 = H / 2, W1 = W / 2;
    const int H2 = H / 4, W2 = W / 4;
    const size_t HW0 = (size_t)H * W, HW1 = (size_t)H1 * W1, HW2 = (size_t)H2 * W2;

    const float* x    = (const float*)d_in[0];
    const float* flow = (const float*)d_in[1];
    const float* t_p  = (const float*)d_in[2];
    const float* g_p  = (const float*)d_in[3];
    const int*   cH_p = (const int*)d_in[4];
    const int*   iH_p = (const int*)d_in[5];

    // param leaf indices: alphabetical jax tree flatten of the params dict.
    // params: fp(3 blk2), gn{down(6),fb,fw,h0b,h0w,h1b,h1w,h2b,h2w,lat0(5),lat1(5),lat2(5),up(6)}, vnet
    // blk2 leaf order: a1,a2,b1,b2,w1,w2
    const int FP0 = 6, FP1 = 12, FP2 = 18;
    const int GN = 24;                       // down blocks at GN+6*i
    const int FB = GN + 36, FW = GN + 37;
    const int H0B = GN + 38, H0W = GN + 39, H1B = GN + 40, H1W = GN + 41, H2B = GN + 42, H2W = GN + 43;
    const int LAT0 = GN + 44, LAT1 = LAT0 + 30, LAT2 = LAT1 + 30, UP = LAT2 + 30, VN = UP + 36;
    // vnet leaves: a1,a2,b1,b2,b3,w1,w2,w3

    struct Blk { const float *a1, *a2, *b1, *b2, *w1, *w2; };
    auto blk = [&](int base) { Blk q;
        q.a1 = (const float*)d_in[base + 0]; q.a2 = (const float*)d_in[base + 1];
        q.b1 = (const float*)d_in[base + 2]; q.b2 = (const float*)d_in[base + 3];
        q.w1 = (const float*)d_in[base + 4]; q.w2 = (const float*)d_in[base + 5]; return q; };

    size_t off = 0;
    auto alloc = [&](size_t elems) -> float* {
        float* p = (float*)((char*)d_ws + off);
        off += (elems * sizeof(float) + 255) & ~(size_t)255;
        return p; };

    auto conv = [&](const float* in, const float* w, const float* b, const float* res,
                    const float* aPre, const float* aPost, float* out,
                    int N, int Cin, int Hin, int Win, int Cout, int Hout, int Wout, int s) {
        dim3 grid((unsigned)(((Wout + 15) / 16) * ((Hout + 3) / 4) * N),
                  (unsigned)((Cout + COUT_TILE - 1) / COUT_TILE));
        if (s == 1)
            k_conv3x3_wmma<1><<<grid, CONV_TPB, 0, stream>>>(in, w, b, res, aPre, aPost, out,
                                                             N, Cin, Hin, Win, Cout, Hout, Wout);
        else
            k_conv3x3_wmma<2><<<grid, CONV_TPB, 0, stream>>>(in, w, b, res, aPre, aPost, out,
                                                             N, Cin, Hin, Win, Cout, Hout, Wout);
    };
    auto ew = [&](size_t n) { return dim3((unsigned)((n + 255) / 256)); };
    auto swp = [](float*& a, float*& b) { float* t = a; a = b; b = t; };

    // ---- input frames (scaled + raw) ----
    float* frames = alloc((size_t)N2 * 3 * HW0);
    float* rawf   = alloc((size_t)N2 * 3 * HW0);
    k_make_frames<<<ew((size_t)N2 * 3 * HW0), 256, 0, stream>>>(x, frames, rawf, B, H, W);

    // ---- feature pyramid ----
    float* t32  = alloc((size_t)N2 * 32 * HW0);
    float* pyr1 = alloc((size_t)N2 * 32 * HW0);
    { Blk q = blk(FP0);
      conv(frames, q.w1, q.b1, nullptr, nullptr, q.a1, t32,  N2, 3,  H, W, 32, H, W, 1);
      conv(t32,    q.w2, q.b2, nullptr, nullptr, q.a2, pyr1, N2, 32, H, W, 32, H, W, 1); }
    float* t64  = alloc((size_t)N2 * 64 * HW1);
    float* pyr2 = alloc((size_t)N2 * 64 * HW1);
    { Blk q = blk(FP1);
      conv(pyr1, q.w1, q.b1, nullptr, nullptr, q.a1, t64,  N2, 32, H,  W,  64, H1, W1, 2);
      conv(t64,  q.w2, q.b2, nullptr, nullptr, q.a2, pyr2, N2, 64, H1, W1, 64, H1, W1, 1); }
    float* t96  = alloc((size_t)N2 * 96 * HW2);
    float* pyr3 = alloc((size_t)N2 * 96 * HW2);
    { Blk q = blk(FP2);
      conv(pyr2, q.w1, q.b1, nullptr, nullptr, q.a1, t96,  N2, 64, H1, W1, 96, H2, W2, 2);
      conv(t96,  q.w2, q.b2, nullptr, nullptr, q.a2, pyr3, N2, 96, H2, W2, 96, H2, W2, 1); }

    // ---- vnet: occlusion z ----
    float* vin = alloc((size_t)N2 * 6 * HW0);
    k_backwarp_bd<<<ew((size_t)N2 * 3 * HW0), 256, 0, stream>>>(frames, flow, vin, B, H, W);
    float* hz1 = alloc((size_t)N2 * 64 * HW0);
    float* hz2 = alloc((size_t)N2 * 64 * HW0);
    float* zb  = alloc((size_t)N2 * 1 * HW0);
    conv(vin, (const float*)d_in[VN + 5], (const float*)d_in[VN + 2], nullptr, nullptr,
         (const float*)d_in[VN + 0], hz1, N2, 6, H, W, 64, H, W, 1);
    conv(hz1, (const float*)d_in[VN + 6], (const float*)d_in[VN + 3], nullptr, nullptr,
         (const float*)d_in[VN + 1], hz2, N2, 64, H, W, 64, H, W, 1);
    conv(hz2, (const float*)d_in[VN + 7], (const float*)d_in[VN + 4], nullptr, nullptr,
         nullptr, zb, N2, 64, H, W, 1, H, W, 1);

    // ---- per-level softmax splatting ----
    float* warped0 = alloc((size_t)N2 * 3  * HW0);
    float* warped1 = alloc((size_t)N2 * 32 * HW0);
    float* warped2 = alloc((size_t)N2 * 64 * HW1);
    float* warped3 = alloc((size_t)N2 * 96 * HW2);
    float* flowt   = alloc((size_t)N2 * 2  * HW0);
    float* accb    = alloc((size_t)N2 * 33 * HW0);   // max (C+1) over levels
    float* flow_s  = alloc((size_t)N2 * 2  * HW1);
    float* z_s     = alloc((size_t)N2 * 1  * HW1);

    auto dolevel = [&](const float* feat, int C, int h, int w, float sc,
                       const float* flv, const float* zlv, float* warped) {
        const size_t hw = (size_t)h * w;
        k_interflow<<<ew((size_t)N2 * 2 * hw), 256, 0, stream>>>(flv, t_p, g_p, cH_p, iH_p,
                                                                 sc, B, h, w, flowt);
        const size_t an = (size_t)N2 * (C + 1) * hw;
        k_zero<<<ew(an), 256, 0, stream>>>(accb, an);
        k_softsplat_scatter<<<ew((size_t)N2 * hw), 256, 0, stream>>>(feat, flowt, zlv, accb, N2, C, h, w);
        k_softsplat_norm<<<ew((size_t)N2 * C * hw), 256, 0, stream>>>(accb, warped, N2, C, h, w);
    };

    dolevel(frames, 3, H, W, 1.0f, flow, zb, warped0);
    {   // warped_img (second output), reuses lv0 flowt
        const size_t an = (size_t)N2 * 4 * HW0;
        k_zero<<<ew(an), 256, 0, stream>>>(accb, an);
        k_softsplat_scatter<<<ew((size_t)N2 * HW0), 256, 0, stream>>>(rawf, flowt, zb, accb, N2, 3, H, W);
        float* wimg = (float*)d_out + (size_t)B * 3 * HW0;
        k_softsplat_norm<<<ew((size_t)N2 * 3 * HW0), 256, 0, stream>>>(accb, wimg, N2, 3, H, W);
    }
    dolevel(pyr1, 32, H, W, 1.0f, flow, zb, warped1);
    k_resize_bilinear<<<ew((size_t)N2 * 2 * HW1), 256, 0, stream>>>(flow, flow_s, N2, 2, H, W, H1, W1, 0.5f);
    k_resize_bilinear<<<ew((size_t)N2 * 1 * HW1), 256, 0, stream>>>(zb, z_s, N2, 1, H, W, H1, W1, 1.0f);
    dolevel(pyr2, 64, H1, W1, 0.5f, flow_s, z_s, warped2);
    k_resize_bilinear<<<ew((size_t)N2 * 2 * HW2), 256, 0, stream>>>(flow, flow_s, N2, 2, H, W, H2, W2, 0.25f);
    k_resize_bilinear<<<ew((size_t)N2 * 1 * HW2), 256, 0, stream>>>(zb, z_s, N2, 1, H, W, H2, W2, 1.0f);
    dolevel(pyr3, 96, H2, W2, 0.25f, flow_s, z_s, warped3);

    // ---- assemble gridnet inputs (channel-slab D2D copies) ----
    float* cat01 = alloc((size_t)2 * 70  * HW0);
    float* f2c   = alloc((size_t)2 * 128 * HW1);
    float* f3c   = alloc((size_t)2 * 192 * HW2);
    auto cpy = [&](float* dst, const float* src, size_t elems) {
        (void)hipMemcpyAsync(dst, src, elems * sizeof(float), hipMemcpyDeviceToDevice, stream); };
    for (int n = 0; n < 2; ++n) {
        float* d0 = cat01 + (size_t)n * 70 * HW0;
        cpy(d0,              warped0 + (size_t)n       * 3  * HW0, 3 * HW0);
        cpy(d0 + 3 * HW0,    warped0 + (size_t)(n + 2) * 3  * HW0, 3 * HW0);
        cpy(d0 + 6 * HW0,    warped1 + (size_t)n       * 32 * HW0, 32 * HW0);
        cpy(d0 + 38 * HW0,   warped1 + (size_t)(n + 2) * 32 * HW0, 32 * HW0);
        float* d2 = f2c + (size_t)n * 128 * HW1;
        cpy(d2,              warped2 + (size_t)n       * 64 * HW1, 64 * HW1);
        cpy(d2 + 64 * HW1,   warped2 + (size_t)(n + 2) * 64 * HW1, 64 * HW1);
        float* d3 = f3c + (size_t)n * 192 * HW2;
        cpy(d3,              warped3 + (size_t)n       * 96 * HW2, 96 * HW2);
        cpy(d3 + 96 * HW2,   warped3 + (size_t)(n + 2) * 96 * HW2, 96 * HW2);
    }

    // ---- gridnet ----
    const int b2 = 2;
    float* r0 = alloc((size_t)b2 * 32 * HW0); float* r0n = alloc((size_t)b2 * 32 * HW0);
    float* r1 = alloc((size_t)b2 * 64 * HW1); float* r1n = alloc((size_t)b2 * 64 * HW1);
    float* r2 = alloc((size_t)b2 * 96 * HW2); float* r2n = alloc((size_t)b2 * 96 * HW2);
    float* t0  = alloc((size_t)b2 * 64 * HW0);   // level-0 resize buffer (64ch)
    float* t0b = alloc((size_t)b2 * 32 * HW0);
    float* t0c = alloc((size_t)b2 * 32 * HW0);
    float* t0d = alloc((size_t)b2 * 32 * HW0);
    float* t1  = alloc((size_t)b2 * 96 * HW1);   // level-1 resize buffer (96ch)
    float* t1b = alloc((size_t)b2 * 64 * HW1);
    float* t1c = alloc((size_t)b2 * 64 * HW1);
    float* t1d = alloc((size_t)b2 * 64 * HW1);
    float* t2a = alloc((size_t)b2 * 96 * HW2);
    float* t2b = alloc((size_t)b2 * 96 * HW2);
    float* t2c = alloc((size_t)b2 * 96 * HW2);

    // heads + initial down chain
    conv(cat01, (const float*)d_in[H0W], (const float*)d_in[H0B], nullptr, nullptr, nullptr,
         r0, b2, 70, H, W, 32, H, W, 1);
    conv(f2c, (const float*)d_in[H1W], (const float*)d_in[H1B], nullptr, nullptr, nullptr,
         t1c, b2, 128, H1, W1, 64, H1, W1, 1);
    { Blk q = blk(GN + 0);
      conv(r0,  q.w1, q.b1, nullptr, q.a1, nullptr, t1d, b2, 32, H,  W,  64, H1, W1, 2);
      conv(t1d, q.w2, q.b2, t1c,     q.a2, nullptr, r1,  b2, 64, H1, W1, 64, H1, W1, 1); }
    conv(f3c, (const float*)d_in[H2W], (const float*)d_in[H2B], nullptr, nullptr, nullptr,
         t2b, b2, 192, H2, W2, 96, H2, W2, 1);
    { Blk q = blk(GN + 6);
      conv(r1,  q.w1, q.b1, nullptr, q.a1, nullptr, t2c, b2, 64, H1, W1, 96, H2, W2, 2);
      conv(t2c, q.w2, q.b2, t2b,     q.a2, nullptr, r2,  b2, 96, H2, W2, 96, H2, W2, 1); }

    auto lat = [&](const float* in, Blk q, float* tmp, float* out, int C, int h, int w) {
        conv(in,  q.w1, q.b1, nullptr, q.a1, nullptr, tmp, b2, C, h, w, C, h, w, 1);
        conv(tmp, q.w2, q.b2, in,      q.a2, nullptr, out, b2, C, h, w, C, h, w, 1);
    };

    for (int i = 0; i < 2; ++i) {   // down phase
        lat(r0, blk(LAT0 + 6 * i), t0b, r0n, 32, H, W);
        lat(r1, blk(LAT1 + 6 * i), t1b, t1c, 64, H1, W1);
        { Blk q = blk(GN + 6 * (2 + 2 * i));
          conv(r0n, q.w1, q.b1, nullptr, q.a1, nullptr, t1d, b2, 32, H,  W,  64, H1, W1, 2);
          conv(t1d, q.w2, q.b2, t1c,     q.a2, nullptr, r1n, b2, 64, H1, W1, 64, H1, W1, 1); }
        lat(r2, blk(LAT2 + 6 * i), t2a, t2b, 96, H2, W2);
        { Blk q = blk(GN + 6 * (3 + 2 * i));
          conv(r1n, q.w1, q.b1, nullptr, q.a1, nullptr, t2c, b2, 64, H1, W1, 96, H2, W2, 2);
          conv(t2c, q.w2, q.b2, t2b,     q.a2, nullptr, r2n, b2, 96, H2, W2, 96, H2, W2, 1); }
        swp(r0, r0n); swp(r1, r1n); swp(r2, r2n);
    }
    for (int i = 2; i < 5; ++i) {   // up phase
        lat(r2, blk(LAT2 + 6 * i), t2a, r2n, 96, H2, W2);
        lat(r1, blk(LAT1 + 6 * i), t1b, t1c, 64, H1, W1);
        k_resize_bilinear<<<ew((size_t)b2 * 96 * HW1), 256, 0, stream>>>(r2n, t1, b2, 96, H2, W2, H1, W1, 1.0f);
        { Blk q = blk(UP + 6 * (2 * (i - 2)));
          conv(t1,  q.w1, q.b1, nullptr, q.a1, nullptr, t1d, b2, 96, H1, W1, 64, H1, W1, 1);
          conv(t1d, q.w2, q.b2, t1c,     q.a2, nullptr, r1n, b2, 64, H1, W1, 64, H1, W1, 1); }
        lat(r0, blk(LAT0 + 6 * i), t0b, t0c, 32, H, W);
        k_resize_bilinear<<<ew((size_t)b2 * 64 * HW0), 256, 0, stream>>>(r1n, t0, b2, 64, H1, W1, H, W, 1.0f);
        { Blk q = blk(UP + 6 * (2 * (i - 2) + 1));
          conv(t0,  q.w1, q.b1, nullptr, q.a1, nullptr, t0d, b2, 64, H, W, 32, H, W, 1);
          conv(t0d, q.w2, q.b2, t0c,     q.a2, nullptr, r0n, b2, 32, H, W, 32, H, W, 1); }
        swp(r0, r0n); swp(r1, r1n); swp(r2, r2n);
    }

    // final conv + clip into first output slot
    conv(r0, (const float*)d_in[FW], (const float*)d_in[FB], nullptr, nullptr, nullptr,
         t0b, b2, 32, H, W, 3, H, W, 1);
    k_clip01<<<ew((size_t)b2 * 3 * HW0), 256, 0, stream>>>(t0b, (float*)d_out, (size_t)b2 * 3 * HW0);
}
